// CriterionLP_84447646974568
// MI455X (gfx1250) — matrix operand order repaired and checked by reading, not compiled, not run on hardware
//
#include <hip/hip_runtime.h>

typedef __bf16 v16bf __attribute__((ext_vector_type(16)));
typedef float  v8f   __attribute__((ext_vector_type(8)));

#define C_DIM   256   // feature dim
#define GSIZE   128   // topk * num_instances (columns per label block)
#define BM      128   // row tile
#define BK      32    // K chunk (WMMA K depth)
#define KP      40    // padded LDS row pitch (bf16 elems): 80B, 16B-aligned rows
#define INV_T   20.0f // 1 / TEMP
#define EPSF    1e-6f

struct U32x8 { uint4 a, b; };

// f32x2 -> packed bf16x2, round-half-up, single v_perm_b32 for the pack.
__device__ __forceinline__ unsigned int pk2bf(float x, float y) {
    unsigned ux = __float_as_uint(x) + 0x8000u;
    unsigned uy = __float_as_uint(y) + 0x8000u;
    // result bytes [1:0] = ux bytes [3:2], bytes [3:2] = uy bytes [3:2]
    return __builtin_amdgcn_perm(uy, ux, 0x07060302u);
}

// Load 16 bf16 (two 16-byte chunks) from LDS into a WMMA fragment.
__device__ __forceinline__ v16bf ld_frag(const unsigned short* base, int e0, int e1) {
    U32x8 t;
    t.a = *(const uint4*)(const void*)(base + e0);
    t.b = *(const uint4*)(const void*)(base + e1);
    return __builtin_bit_cast(v16bf, t);
}

// One workgroup: 128 rows x one full 128-col label block. GEMM (bf16 WMMA,
// f32 acc) fused with exp() and per-row block min/max.
__global__ __launch_bounds__(256) void lp_blockstats_kernel(
    const float* __restrict__ feats, const float* __restrict__ feats_s,
    float* __restrict__ blk_min, float* __restrict__ blk_max, int Kgroups)
{
    __shared__ __align__(16) unsigned short sA[BM * KP];
    __shared__ __align__(16) unsigned short sB[BM * KP];

    const int tid  = threadIdx.x;
    const int lane = tid & 31;
    const int wave = tid >> 5;
    const int wm   = wave & 3;    // 4 row groups of 32
    const int wn   = wave >> 2;   // 2 col groups of 64
    const int half = lane >> 4;
    const int l16  = lane & 15;

    const int grp     = blockIdx.x;        // label block (column tile)
    const int rowBase = blockIdx.y * BM;

    // Loop-invariant staging addresses (idx = tid + it*256 -> row, quad)
    const int srow0 = tid >> 3;            // rows for it = 0..3: srow0 + it*32
    const int sq    = tid & 7;
    const float* ApBase = feats   + (size_t)rowBase * C_DIM       + srow0 * C_DIM + sq * 4;
    const float* BpBase = feats_s + (size_t)grp * GSIZE * C_DIM   + srow0 * C_DIM + sq * 4;

    // Loop-invariant fragment LDS element offsets
    int aoff[2], boff[4];
#pragma unroll
    for (int i = 0; i < 2; ++i)
        aoff[i] = (wm * 32 + i * 16 + l16) * KP + 8 * half;   // + (e0, e0+16)
#pragma unroll
    for (int j = 0; j < 4; ++j)
        boff[j] = (wn * 64 + j * 16 + l16) * KP + 8 * half;   // + (e0, e0+16)

    v8f acc[2][4];
#pragma unroll
    for (int i = 0; i < 2; ++i)
#pragma unroll
        for (int j = 0; j < 4; ++j)
            acc[i][j] = (v8f){0.f,0.f,0.f,0.f,0.f,0.f,0.f,0.f};

#pragma unroll
    for (int kc = 0; kc < C_DIM; kc += BK) {
        // ---- stage f32 -> bf16 into LDS (128 rows x 32 k) ----
#pragma unroll
        for (int it = 0; it < 4; ++it) {
            const int row = srow0 + it * 32;
            float4 av = *(const float4*)(const void*)(ApBase + it * 32 * C_DIM + kc);
            float4 bv = *(const float4*)(const void*)(BpBase + it * 32 * C_DIM + kc);
            uint2 pa; pa.x = pk2bf(av.x, av.y); pa.y = pk2bf(av.z, av.w);
            uint2 pb; pb.x = pk2bf(bv.x, bv.y); pb.y = pk2bf(bv.z, bv.w);
            *(uint2*)(void*)&sA[row * KP + sq * 4] = pa;
            *(uint2*)(void*)&sB[row * KP + sq * 4] = pb;
            if (kc + BK < C_DIM) {  // global_prefetch_b8 for next chunk
                __builtin_prefetch(ApBase + it * 32 * C_DIM + kc + BK, 0, 1);
                __builtin_prefetch(BpBase + it * 32 * C_DIM + kc + BK, 0, 1);
            }
        }
        __syncthreads();

        // ---- fragments per ISA VGPR layout ----
        v16bf afr[2], bfr[4];
#pragma unroll
        for (int i = 0; i < 2; ++i)
            afr[i] = ld_frag(sA, aoff[i], aoff[i] + 16);   // K chunks 8*half, 16+8*half
#pragma unroll
        for (int j = 0; j < 4; ++j)
            bfr[j] = ld_frag(sB, boff[j], boff[j] + 16);   // K chunks 8*half, 16+8*half

#pragma unroll
        for (int i = 0; i < 2; ++i)
#pragma unroll
            for (int j = 0; j < 4; ++j)
                acc[i][j] = __builtin_amdgcn_wmma_f32_16x16x32_bf16(
                    false, afr[i], false, bfr[j], (short)0, acc[i][j], false, false);
        __syncthreads();
    }

    // ---- fused epilogue: exp, per-row min/max over this 128-col block ----
    float* pmn = (float*)sA;   // [128][2] partials (LDS reuse, post-barrier)
    float* pmx = (float*)sB;

    union V8 { v8f v; float f[8]; };
#pragma unroll
    for (int i = 0; i < 2; ++i) {
        V8 u0, u1, u2, u3;
        u0.v = acc[i][0]; u1.v = acc[i][1]; u2.v = acc[i][2]; u3.v = acc[i][3];
#pragma unroll
        for (int v = 0; v < 8; ++v) {
            float e0 = __expf(u0.f[v] * INV_T);
            float e1 = __expf(u1.f[v] * INV_T);
            float e2 = __expf(u2.f[v] * INV_T);
            float e3 = __expf(u3.f[v] * INV_T);
            float mn = fminf(fminf(e0, e1), fminf(e2, e3));
            float mx = fmaxf(fmaxf(e0, e1), fmaxf(e2, e3));
#pragma unroll
            for (int off = 1; off < 16; off <<= 1) {  // reduce 16-lane half
                mn = fminf(mn, __shfl_xor(mn, off));
                mx = fmaxf(mx, __shfl_xor(mx, off));
            }
            if (l16 == 0) {  // C-layout: row = 8*half + v within 16-tile
                int lr = wm * 32 + i * 16 + 8 * half + v;
                pmn[lr * 2 + wn] = mn;
                pmx[lr * 2 + wn] = mx;
            }
        }
    }
    __syncthreads();

    if (tid < BM) {
        float mn = fminf(pmn[tid * 2 + 0], pmn[tid * 2 + 1]);
        float mx = fmaxf(pmx[tid * 2 + 0], pmx[tid * 2 + 1]);
        size_t o = (size_t)(rowBase + tid) * Kgroups + grp;
        blk_min[o] = mn;
        blk_max[o] = mx;
    }
}

// Per-row masked pos/neg sums + -log
__global__ __launch_bounds__(256) void lp_rowloss_kernel(
    const float* __restrict__ blk_min, const float* __restrict__ blk_max,
    const int* __restrict__ labels, const int* __restrict__ labels_s,
    float* __restrict__ loss_rows, int B, int Kgroups)
{
    int r = blockIdx.x * blockDim.x + threadIdx.x;
    if (r >= B) return;
    int lab = labels[r];
    float pos = 0.f, neg = 0.f;
    for (int k = 0; k < Kgroups; ++k) {
        int gl   = labels_s[k * GSIZE];
        float mn = blk_min[(size_t)r * Kgroups + k];
        float mx = blk_max[(size_t)r * Kgroups + k];
        if (gl == lab) pos += mn; else neg += mx;
    }
    loss_rows[r] = -__logf(pos / (pos + neg + EPSF) + EPSF);
}

// Mean over rows -> scalar
__global__ __launch_bounds__(256) void lp_mean_kernel(
    const float* __restrict__ loss_rows, float* __restrict__ out, int B)
{
    __shared__ float sred[256];
    float s = 0.f;
    for (int i = threadIdx.x; i < B; i += 256) s += loss_rows[i];
    sred[threadIdx.x] = s;
    __syncthreads();
    for (int st = 128; st > 0; st >>= 1) {
        if (threadIdx.x < st) sred[threadIdx.x] += sred[threadIdx.x + st];
        __syncthreads();
    }
    if (threadIdx.x == 0) out[0] = sred[0] / (float)B;
}

extern "C" void kernel_launch(void* const* d_in, const int* in_sizes, int n_in,
                              void* d_out, int out_size, void* d_ws, size_t ws_size,
                              hipStream_t stream) {
    const float* feats    = (const float*)d_in[0];
    const float* feats_s  = (const float*)d_in[1];
    const int*   labels   = (const int*)d_in[2];
    const int*   labels_s = (const int*)d_in[3];

    const int B  = in_sizes[0] / C_DIM;          // 2048
    const int M  = in_sizes[1] / C_DIM;          // 16384
    const int Kg = M / GSIZE;                    // 128 label blocks

    float* blk_min   = (float*)d_ws;
    float* blk_max   = blk_min + (size_t)B * Kg;
    float* loss_rows = blk_max + (size_t)B * Kg;

    dim3 grid(Kg, B / BM);
    lp_blockstats_kernel<<<grid, 256, 0, stream>>>(feats, feats_s, blk_min, blk_max, Kg);
    lp_rowloss_kernel<<<dim3((B + 255) / 256), 256, 0, stream>>>(
        blk_min, blk_max, labels, labels_s, loss_rows, B, Kg);
    lp_mean_kernel<<<1, 256, 0, stream>>>(loss_rows, (float*)d_out, B);
}
